// SelfAttention_89386859364653
// MI455X (gfx1250) — matrix operand order repaired and checked by reading, not compiled, not run on hardware
//
#include <hip/hip_runtime.h>

typedef __attribute__((ext_vector_type(16))) _Float16 v16h;
typedef __attribute__((ext_vector_type(8)))  _Float16 v8h;
typedef __attribute__((ext_vector_type(4)))  _Float16 v4h;
typedef __attribute__((ext_vector_type(8)))  float    v8f;
typedef __attribute__((ext_vector_type(4)))  float    v4f;

#define BATCH 16
#define SEQ   2048
#define DKDIM 128
#define KC    32              // keys per staged chunk
#define NCHUNK (SEQ / KC)
#define NWAVE 8               // waves per block (wave32)
#define QBLK  (NWAVE * 16)    // query rows per block

// Butterfly max across a 16-lane DPP row: xor1, xor2, half_mirror, mirror.
__device__ __forceinline__ float dpp_fmax_x16(float x) {
  int t;
  t = __builtin_amdgcn_update_dpp(0, __builtin_bit_cast(int, x),
                                  0xB1, 0xF, 0xF, true);   // quad_perm xor1
  x = fmaxf(x, __builtin_bit_cast(float, t));
  t = __builtin_amdgcn_update_dpp(0, __builtin_bit_cast(int, x),
                                  0x4E, 0xF, 0xF, true);   // quad_perm xor2
  x = fmaxf(x, __builtin_bit_cast(float, t));
  t = __builtin_amdgcn_update_dpp(0, __builtin_bit_cast(int, x),
                                  0x141, 0xF, 0xF, true);  // row_half_mirror
  x = fmaxf(x, __builtin_bit_cast(float, t));
  t = __builtin_amdgcn_update_dpp(0, __builtin_bit_cast(int, x),
                                  0x140, 0xF, 0xF, true);  // row_mirror
  x = fmaxf(x, __builtin_bit_cast(float, t));
  return x;
}

// Issue global loads for one K/V chunk into registers (no waits here).
__device__ __forceinline__ void stage_load(const float* __restrict__ K,
                                           const float* __restrict__ V,
                                           int kc0, int tid,
                                           v4f (&kreg)[4], v4f (&vreg)[4]) {
#pragma unroll
  for (int it = 0; it < 4; ++it) {           // K: contiguous float4 per thread
    const int i4 = tid * 4 + it * 1024;
    const int kr = i4 >> 7;
    const int kd = i4 & (DKDIM - 1);
    kreg[it] = *(const v4f*)&K[(size_t)(kc0 + kr) * DKDIM + kd];
  }
#pragma unroll
  for (int it = 0; it < 4; ++it) {           // V: 4 keys per fixed dk column
    const int j   = tid + it * 256;
    const int kd  = j & (DKDIM - 1);
    const int krb = (j >> 7) << 2;
#pragma unroll
    for (int u = 0; u < 4; ++u)
      vreg[it][u] = V[(size_t)(kc0 + krb + u) * DKDIM + kd];
  }
}

// Convert staged registers to f16 and store into an LDS buffer pair.
__device__ __forceinline__ void stage_store(_Float16* sKbuf, _Float16* sVtbuf,
                                            int tid,
                                            const v4f (&kreg)[4],
                                            const v4f (&vreg)[4]) {
#pragma unroll
  for (int it = 0; it < 4; ++it) {
    const int i4 = tid * 4 + it * 1024;
    const int kr = i4 >> 7;
    const int kd = i4 & (DKDIM - 1);
    v4h kh = { (_Float16)kreg[it][0], (_Float16)kreg[it][1],
               (_Float16)kreg[it][2], (_Float16)kreg[it][3] };
    *(v4h*)&sKbuf[kr * DKDIM + kd] = kh;
  }
#pragma unroll
  for (int it = 0; it < 4; ++it) {
    const int j   = tid + it * 256;
    const int kd  = j & (DKDIM - 1);
    const int krb = (j >> 7) << 2;
    v4h vh = { (_Float16)vreg[it][0], (_Float16)vreg[it][1],
               (_Float16)vreg[it][2], (_Float16)vreg[it][3] };
    *(v4h*)&sVtbuf[kd * KC + krb] = vh;
  }
}

__global__ __launch_bounds__(256)
void fa_fwd_kernel(const float* __restrict__ qg,
                   const float* __restrict__ kg,
                   const float* __restrict__ vg,
                   float* __restrict__ og) {
  // double-buffered K/V staging + P staging: 40KB total
  __shared__ __align__(16) _Float16 sK [2][KC * DKDIM];
  __shared__ __align__(16) _Float16 sVt[2][DKDIM * KC];
  __shared__ __align__(16) _Float16 sP [NWAVE * 16 * KC];

  const int tid   = threadIdx.x;
  const int wave  = tid >> 5;
  const int lane  = tid & 31;
  const int hlane = lane & 15;
  const int half  = lane >> 4;      // 0: lanes 0-15, 1: lanes 16-31
  const int abase = half * 8;       // A-frag K interleave base (0 / 8)
  const int bbase = half * 16;      // B-frag K split base (0 / 16)

  const int b  = blockIdx.y;
  const int q0 = blockIdx.x * QBLK + wave * 16;

  const float scale = 0.08838834764831845f;   // 1/sqrt(128)

  const float* Q = qg + ((size_t)b * SEQ + q0) * DKDIM;
  const float* K = kg + (size_t)b * SEQ * DKDIM;
  const float* V = vg + (size_t)b * SEQ * DKDIM;

  // ---- Q tile -> 4 A fragments (f32 -> f16, pre-scaled), held in VGPRs ----
  v16h aq[4];
  {
    const float* qr = Q + hlane * DKDIM;
#pragma unroll
    for (int c = 0; c < 4; ++c) {
#pragma unroll
      for (int e = 0; e < 16; ++e) {
        const int kk = c * 32 + bbase + e;
        aq[c][e] = (_Float16)(qr[kk] * scale);
      }
    }
  }

  // constant all-ones B fragment: row-sum of P via the matrix pipe
  v16h onesf;
#pragma unroll
  for (int e = 0; e < 16; ++e) onesf[e] = (_Float16)1.0f;

  v8f acc[8] = {};          // 16 rows x 128 cols of O, fp32 (C layout)
  v8f lacc = {};            // running softmax denominator (same C layout)
  float mrow[8];
#pragma unroll
  for (int r = 0; r < 8; ++r) mrow[r] = -3.0e38f;

  _Float16* myP = sP + wave * (16 * KC);

  // ---- prologue: stage chunk 0 ----
  v4f kreg[4], vreg[4];
  stage_load(K, V, 0, tid, kreg, vreg);
  stage_store(sK[0], sVt[0], tid, kreg, vreg);
  __syncthreads();

  for (int ic = 0; ic < NCHUNK; ++ic) {
    const int cur  = ic & 1;
    const int nxt  = cur ^ 1;
    const bool more = (ic + 1) < NCHUNK;
    const _Float16* cK  = sK[cur];
    const _Float16* cVt = sVt[cur];

    // issue next chunk's global loads now; latency hides behind the WMMAs
    if (more) stage_load(K, V, (ic + 1) * KC, tid, kreg, vreg);

    // ---- scores: S = Q * K^T, two 16-key tiles, accumulate over DK ----
    v16h bk[8];
#pragma unroll
    for (int c = 0; c < 4; ++c) {
      bk[c]     = *(const v16h*)&cK[(hlane     ) * DKDIM + c * 32 + bbase];
      bk[4 + c] = *(const v16h*)&cK[(hlane + 16) * DKDIM + c * 32 + bbase];
    }
    v8f s0 = {}, s1 = {};
#pragma unroll
    for (int c = 0; c < 4; ++c) {
      s0 = __builtin_amdgcn_wmma_f32_16x16x32_f16(false, aq[c], false, bk[c],
                                                  (short)0, s0, false, false);
      s1 = __builtin_amdgcn_wmma_f32_16x16x32_f16(false, aq[c], false, bk[4 + c],
                                                  (short)0, s1, false, false);
    }

    // ---- online softmax: DPP row max, exp, rescale ----
#pragma unroll
    for (int r = 0; r < 8; ++r) {
      const float tmax  = dpp_fmax_x16(fmaxf(s0[r], s1[r]));
      const float mnew  = fmaxf(mrow[r], tmax);
      const float alpha = __expf(mrow[r] - mnew);
      mrow[r] = mnew;
      s0[r] = __expf(s0[r] - mnew);
      s1[r] = __expf(s1[r] - mnew);
      lacc[r] *= alpha;
#pragma unroll
      for (int t = 0; t < 8; ++t) acc[t][r] *= alpha;
    }

    // ---- P (C layout) -> per-wave LDS (row-major 16 x 32, f16) ----
    {
      const int pr = half * 8;
#pragma unroll
      for (int r = 0; r < 8; ++r) {
        myP[(pr + r) * KC + hlane]      = (_Float16)s0[r];
        myP[(pr + r) * KC + 16 + hlane] = (_Float16)s1[r];
      }
    }
    // per-wave region; DS ops are in-order within a wave, no barrier needed

    // ---- PV: A = P (16x32); B = Vt tiles + ones (row-sum) ----
    v16h ap;
    {
      const _Float16* prow = myP + hlane * KC;
      const v8h lo = *(const v8h*)(prow + bbase);        // K = bbase..bbase+7
      const v8h hi = *(const v8h*)(prow + bbase + 8);    // K = bbase+8..+15
      ap = __builtin_shufflevector(lo, hi,
                                   0, 1, 2, 3, 4, 5, 6, 7,
                                   8, 9, 10, 11, 12, 13, 14, 15);
    }
    v16h bv[8];
#pragma unroll
    for (int t = 0; t < 8; ++t)
      bv[t] = *(const v16h*)&cVt[(t * 16 + hlane) * KC + bbase];

    lacc = __builtin_amdgcn_wmma_f32_16x16x32_f16(false, ap, false, onesf,
                                                  (short)0, lacc, false, false);
#pragma unroll
    for (int t = 0; t < 8; ++t)
      acc[t] = __builtin_amdgcn_wmma_f32_16x16x32_f16(false, ap, false, bv[t],
                                                      (short)0, acc[t],
                                                      false, false);

    // ---- write next chunk into the other buffer, then one barrier ----
    if (more) stage_store(sK[nxt], sVt[nxt], tid, kreg, vreg);
    __syncthreads();
  }

  // ---- epilogue: divide by row sum, store fp32 ----
  float* O = og + ((size_t)b * SEQ + q0) * DKDIM;
  const int orow = half * 8;
#pragma unroll
  for (int r = 0; r < 8; ++r) {
    const float inv = 1.0f / lacc[r];
#pragma unroll
    for (int t = 0; t < 8; ++t) {
      O[(size_t)(orow + r) * DKDIM + t * 16 + hlane] = acc[t][r] * inv;
    }
  }
}

extern "C" void kernel_launch(void* const* d_in, const int* in_sizes, int n_in,
                              void* d_out, int out_size, void* d_ws, size_t ws_size,
                              hipStream_t stream) {
  (void)in_sizes; (void)n_in; (void)out_size; (void)d_ws; (void)ws_size;
  const float* q = (const float*)d_in[0];
  const float* k = (const float*)d_in[1];
  const float* v = (const float*)d_in[2];
  float* o = (float*)d_out;
  dim3 grid(SEQ / QBLK, BATCH);
  fa_fwd_kernel<<<grid, dim3(256), 0, stream>>>(q, k, v, o);
}
